// MuBottleneck_70677981823588
// MI455X (gfx1250) — matrix operand order repaired and checked by reading, not compiled, not run on hardware
//
#include <hip/hip_runtime.h>
#include <hip/hip_bf16.h>

typedef _Float16 half_t;
typedef __attribute__((ext_vector_type(16))) _Float16 v16h;
typedef __attribute__((ext_vector_type(8)))  _Float16 v8h;
typedef __attribute__((ext_vector_type(8)))  float    v8f;

constexpr int Bn  = 4;
constexpr int Cc  = 256;
constexpr int Ll  = 2048;
constexpr int Hh  = 8;
constexpr int DHd = 32;
constexpr int W4  = 1024;
constexpr int C3  = 3 * Cc;      // 768
constexpr int MT  = Bn * Ll;     // 8192 rows of the folded [B*L, *] activations
constexpr float GN_EPS = 1e-5f;

// ---------------------------------------------------------------- helpers
__device__ inline v16h load_frag(const half_t* p0, const half_t* p1) {
  union { v16h v; v8h h[2]; } u;
  u.h[0] = *(const v8h*)p0;
  u.h[1] = *(const v8h*)p1;
  return u.v;
}

// ---------------------------------------------------------------- prep kernels
__global__ void cvt_f32_to_f16(const float* __restrict__ in, half_t* __restrict__ out, int n) {
  int i = blockIdx.x * blockDim.x + threadIdx.x;
  if (i < n) out[i] = (half_t)in[i];
}

// x [B,C,L] f32 -> xT [B,L,C] f16
__global__ void transpose_x_f16(const float* __restrict__ x, half_t* __restrict__ xT) {
  int i = blockIdx.x * blockDim.x + threadIdx.x;       // index into xT, B*L*C total
  int c = i & (Cc - 1);
  int l = (i >> 8) & (Ll - 1);
  int b = i >> 19;
  xT[i] = (half_t)x[((size_t)b * Cc + c) * Ll + l];
}

// qkv [B,L,3C] f16 -> vT [B,H,DH,L] f16  (V part only)
__global__ void transpose_v_f16(const half_t* __restrict__ qkv, half_t* __restrict__ vT) {
  int i = blockIdx.x * blockDim.x + threadIdx.x;       // B*H*DH*L total
  int l = i & (Ll - 1);
  int d = (i >> 11) & (DHd - 1);
  int h = (i >> 16) & (Hh - 1);
  int b = i >> 19;
  vT[i] = qkv[((size_t)(b * Ll + l)) * C3 + 2 * Cc + h * DHd + d];
}

// ---------------------------------------------------------------- generic WMMA GEMM
// out[m, n] = sum_k A[m,k] * W[n,k] (+ bias[n])   A:[M,K] f16 row-major, W:[N,K] f16 row-major
// Block: 128 threads (4 waves). Block tile 128(M) x 64(N); per-wave tile 32 x 64.
// B tile (64 x 32 halves = 4 KB) is staged through LDS with async-to-LDS copies,
// double buffered: tile t+1 is in flight (ASYNCcnt) while tile t feeds the WMMAs.
enum { MODE_F16_BIAS = 0, MODE_F32_BIAS = 1, MODE_GELU_F16 = 2, MODE_RESID_TR = 3 };

__global__ __launch_bounds__(128)
void gemm_wmma(const half_t* __restrict__ A, const half_t* __restrict__ W,
               const float* __restrict__ bias, void* __restrict__ out,
               const float* __restrict__ identity,   // x [B,C,L] for RESID mode
               int M, int N, int K, int mode)
{
  __shared__ half_t bs[2][64 * 32];                  // double-buffered B tile
  const int tid  = threadIdx.x;
  const int wave = tid >> 5;
  const int lane = tid & 31;
  const int g    = lane >> 4;                        // wave32 half-group
  const int ln   = lane & 15;
  const int m0   = blockIdx.y * 128 + wave * 32;
  const int n0   = blockIdx.x * 64;

  // async B-tile stage: 256 chunks of 16B; 2 chunks per thread
  auto issue_tile = [&](int k, int buf) {
    unsigned lbase = (unsigned)(uintptr_t)(&bs[buf][0]);   // low 32 bits = LDS offset
    #pragma unroll
    for (int q = 0; q < 2; ++q) {
      int ch = q * 128 + tid;
      int i = ch >> 2;                               // B row 0..63
      int j = ch & 3;                                // 16B sub-chunk
      unsigned long long ga =
          (unsigned long long)(W + (size_t)(n0 + i) * K + k + j * 8);
      unsigned la = lbase + (unsigned)(i * 64 + j * 16);
      asm volatile("global_load_async_to_lds_b128 %0, %1, off"
                   :: "v"(la), "v"(ga) : "memory");
    }
  };

  v8f acc[8] = {};                                   // acc[mt*4 + s]
  const half_t* arow0 = A + (size_t)(m0 + ln) * K + g * 8;
  const half_t* arow1 = arow0 + (size_t)16 * K;

  issue_tile(0, 0);
  const int KT = K >> 5;
  for (int kt = 0; kt < KT; ++kt) {
    const int k = kt << 5;
    const int buf = kt & 1;
    if (kt + 1 < KT) {
      issue_tile(k + 32, buf ^ 1);
      asm volatile("s_wait_asynccnt 0x2" ::: "memory");  // tile kt landed
    } else {
      asm volatile("s_wait_asynccnt 0x0" ::: "memory");
    }
    __syncthreads();

    __builtin_prefetch(arow0 + k + 256, 0, 1);       // global_prefetch_b8 on A stream
    v16h a0 = load_frag(arow0 + k, arow0 + k + 16);
    v16h a1 = load_frag(arow1 + k, arow1 + k + 16);
    #pragma unroll
    for (int s = 0; s < 4; ++s) {
      const half_t* bp = &bs[buf][(s * 16 + ln) * 32 + g * 8];
      v16h bf = load_frag(bp, bp + 16);
      acc[s]     = __builtin_amdgcn_wmma_f32_16x16x32_f16(false, a0, false, bf, (short)0, acc[s],     false, false);
      acc[4 + s] = __builtin_amdgcn_wmma_f32_16x16x32_f16(false, a1, false, bf, (short)0, acc[4 + s], false, false);
    }
    __syncthreads();                                 // all waves done reading buf
  }

  #pragma unroll
  for (int mt = 0; mt < 2; ++mt) {
    #pragma unroll
    for (int s = 0; s < 4; ++s) {
      const int col = n0 + s * 16 + ln;
      const float bv = bias ? bias[col] : 0.0f;
      #pragma unroll
      for (int r = 0; r < 8; ++r) {
        const int m = m0 + mt * 16 + g * 8 + r;
        float v = acc[mt * 4 + s][r] + bv;
        if (mode == MODE_F16_BIAS) {
          ((half_t*)out)[(size_t)m * N + col] = (half_t)v;
        } else if (mode == MODE_F32_BIAS) {
          ((float*)out)[(size_t)m * N + col] = v;
        } else if (mode == MODE_GELU_F16) {
          float ge = 0.5f * v * (1.0f + erff(v * 0.70710678118654752f));
          ((half_t*)out)[(size_t)m * N + col] = (half_t)ge;
        } else { // MODE_RESID_TR: out[b][c][l] = x[b][c][l] + v  (m = b*L + l, col = c)
          int b = m >> 11, l = m & (Ll - 1);
          size_t idx = ((size_t)b * Cc + col) * Ll + l;
          ((float*)out)[idx] = identity[idx] + v;
        }
      }
    }
  }
}

// ---------------------------------------------------------------- flash attention
// One wave per (16-row tile, head, batch). qkv: [B,L,3C] f16; vT: [B,H,DH,L] f16.
// Writes attention output (pre out-proj) to attn [B,L,C] f16.
__global__ __launch_bounds__(32)
void attn_flash(const half_t* __restrict__ qkv, const half_t* __restrict__ vT,
                half_t* __restrict__ attn)
{
  __shared__ half_t pl[16 * 40];               // P staging, row stride 40 halves (80B, 16B-multiple)
  const int lane = threadIdx.x & 31;
  const int g = lane >> 4, ln = lane & 15;
  const int m0 = blockIdx.x * 16;
  const int h  = blockIdx.y;
  const int b  = blockIdx.z;
  const float scale = 0.17677669529663687f;    // 1/sqrt(32)

  // Q A-fragment (K = DH = 32 exactly -> one fragment for the whole loop)
  const half_t* qbase = qkv + (size_t)(b * Ll + m0 + ln) * C3 + h * DHd + g * 8;
  v16h qa = load_frag(qbase, qbase + 16);

  v8f o0 = {}, o1 = {};
  float rmax[8], rsum[8];
  #pragma unroll
  for (int r = 0; r < 8; ++r) { rmax[r] = -1e30f; rsum[r] = 0.0f; }

  const half_t* kbase = qkv + (size_t)b * Ll * C3 + Cc + h * DHd + g * 8;
  const half_t* vbase = vT + ((size_t)(b * Hh + h) * DHd) * Ll + g * 8;

  for (int j = 0; j < Ll; j += 32) {
    // S tile: two 16x16 WMMAs (Q[16x32] @ K^T)
    v8f s0, s1;
    {
      const half_t* kp0 = kbase + (size_t)(j + ln) * C3;
      const half_t* kp1 = kbase + (size_t)(j + 16 + ln) * C3;
      v16h kf0 = load_frag(kp0, kp0 + 16);
      v16h kf1 = load_frag(kp1, kp1 + 16);
      v8f z = {};
      s0 = __builtin_amdgcn_wmma_f32_16x16x32_f16(false, qa, false, kf0, (short)0, z, false, false);
      s1 = __builtin_amdgcn_wmma_f32_16x16x32_f16(false, qa, false, kf1, (short)0, z, false, false);
    }
    // online softmax (row reductions across the 16-lane half-groups)
    #pragma unroll
    for (int r = 0; r < 8; ++r) {
      float a0 = s0[r] * scale, a1 = s1[r] * scale;
      float tm = fmaxf(a0, a1);
      tm = fmaxf(tm, __shfl_xor(tm, 1, 32));
      tm = fmaxf(tm, __shfl_xor(tm, 2, 32));
      tm = fmaxf(tm, __shfl_xor(tm, 4, 32));
      tm = fmaxf(tm, __shfl_xor(tm, 8, 32));
      float nm = fmaxf(rmax[r], tm);
      float cr = __expf(rmax[r] - nm);
      rmax[r] = nm;
      float p0 = __expf(a0 - nm), p1 = __expf(a1 - nm);
      float ps = p0 + p1;
      ps += __shfl_xor(ps, 1, 32);
      ps += __shfl_xor(ps, 2, 32);
      ps += __shfl_xor(ps, 4, 32);
      ps += __shfl_xor(ps, 8, 32);
      rsum[r] = rsum[r] * cr + ps;
      o0[r] *= cr;
      o1[r] *= cr;
      // stage P (D layout -> LDS [m][kv])
      pl[(g * 8 + r) * 40 + ln]      = (half_t)p0;
      pl[(g * 8 + r) * 40 + 16 + ln] = (half_t)p1;
    }
    __syncthreads();
    const half_t* pr = &pl[ln * 40 + g * 8];
    v16h pa = load_frag(pr, pr + 16);            // P as A-fragment [16 x 32]
    __syncthreads();

    // O += P @ V : two 16x16 WMMAs over d-subtiles
    const half_t* vp0 = vbase + (size_t)ln * Ll + j;
    const half_t* vp1 = vbase + (size_t)(16 + ln) * Ll + j;
    v16h vf0 = load_frag(vp0, vp0 + 16);
    v16h vf1 = load_frag(vp1, vp1 + 16);
    o0 = __builtin_amdgcn_wmma_f32_16x16x32_f16(false, pa, false, vf0, (short)0, o0, false, false);
    o1 = __builtin_amdgcn_wmma_f32_16x16x32_f16(false, pa, false, vf1, (short)0, o1, false, false);
  }

  #pragma unroll
  for (int r = 0; r < 8; ++r) {
    float inv = 1.0f / rsum[r];
    int m = m0 + g * 8 + r;
    half_t* op = attn + (size_t)(b * Ll + m) * Cc + h * DHd;
    op[ln]      = (half_t)(o0[r] * inv);
    op[16 + ln] = (half_t)(o1[r] * inv);
  }
}

// ---------------------------------------------------------------- GroupNorm(1)
__global__ void gn_zero(float* stats) { if (threadIdx.x < 2 * Bn) stats[threadIdx.x] = 0.0f; }

__global__ __launch_bounds__(256)
void gn_reduce(const float* __restrict__ hm, float* __restrict__ stats) {
  const int b = blockIdx.y;
  const float* p = hm + (size_t)b * (Cc * Ll);
  float s = 0.0f, ss = 0.0f;
  for (int i = blockIdx.x * 256 + threadIdx.x; i < Cc * Ll; i += gridDim.x * 256) {
    float v = p[i]; s += v; ss += v * v;
  }
  __shared__ float sh0[256], sh1[256];
  sh0[threadIdx.x] = s; sh1[threadIdx.x] = ss;
  __syncthreads();
  for (int o = 128; o > 0; o >>= 1) {
    if (threadIdx.x < o) { sh0[threadIdx.x] += sh0[threadIdx.x + o]; sh1[threadIdx.x] += sh1[threadIdx.x + o]; }
    __syncthreads();
  }
  if (threadIdx.x == 0) { atomicAdd(&stats[2 * b], sh0[0]); atomicAdd(&stats[2 * b + 1], sh1[0]); }
}

__global__ void gn_apply(const float* __restrict__ hm, const float* __restrict__ stats,
                         const float* __restrict__ gg, const float* __restrict__ gb,
                         half_t* __restrict__ out) {
  int i = blockIdx.x * blockDim.x + threadIdx.x;   // over B*L*C ([B,L,C] layout)
  int c = i & (Cc - 1);
  int b = i >> 19;
  const float inv_n = 1.0f / (float)(Cc * Ll);
  float mu  = stats[2 * b] * inv_n;
  float var = stats[2 * b + 1] * inv_n - mu * mu;
  float rinv = rsqrtf(var + GN_EPS);
  out[i] = (half_t)((hm[i] - mu) * rinv * gg[c] + gb[c]);
}

// ---------------------------------------------------------------- launcher
extern "C" void kernel_launch(void* const* d_in, const int* in_sizes, int n_in,
                              void* d_out, int out_size, void* d_ws, size_t ws_size,
                              hipStream_t stream) {
  const float* x     = (const float*)d_in[0];
  const float* w_qkv = (const float*)d_in[1];
  const float* b_qkv = (const float*)d_in[2];
  const float* w_o   = (const float*)d_in[3];
  const float* b_o   = (const float*)d_in[4];
  const float* gn_g  = (const float*)d_in[5];
  const float* gn_b  = (const float*)d_in[6];
  const float* w1    = (const float*)d_in[7];
  const float* w2    = (const float*)d_in[8];
  float* out = (float*)d_out;

  char* ws = (char*)d_ws;
  size_t off = 0;
  auto alloc = [&](size_t bytes) { char* p = ws + off; off += (bytes + 255) & ~size_t(255); return p; };
  half_t* xT_h    = (half_t*)alloc((size_t)MT * Cc * 2);       // x transposed [B,L,C]
  half_t* wqkv_h  = (half_t*)alloc((size_t)C3 * Cc * 2);
  half_t* wo_h    = (half_t*)alloc((size_t)Cc * Cc * 2);
  half_t* w1_h    = (half_t*)alloc((size_t)W4 * Cc * 2);
  half_t* w2_h    = (half_t*)alloc((size_t)Cc * W4 * 2);
  half_t* qkv_h   = (half_t*)alloc((size_t)MT * C3 * 2);       // [B,L,3C]
  half_t* vT_h    = (half_t*)alloc((size_t)Bn * Hh * DHd * Ll * 2);
  half_t* attn_h  = (half_t*)alloc((size_t)MT * Cc * 2);       // [B,L,C]
  float*  hmha    = (float*) alloc((size_t)MT * Cc * 4);       // [B,L,C] f32
  half_t* gn_h    = (half_t*)alloc((size_t)MT * Cc * 2);       // [B,L,C]
  half_t* h1g     = (half_t*)alloc((size_t)MT * W4 * 2);       // [B,L,4C] post-GELU
  float*  stats   = (float*) alloc(2 * Bn * 4);

  // 1. precision conversion + layout prep
  cvt_f32_to_f16<<<(C3 * Cc + 255) / 256, 256, 0, stream>>>(w_qkv, wqkv_h, C3 * Cc);
  cvt_f32_to_f16<<<(Cc * Cc + 255) / 256, 256, 0, stream>>>(w_o, wo_h, Cc * Cc);
  cvt_f32_to_f16<<<(W4 * Cc + 255) / 256, 256, 0, stream>>>(w1, w1_h, W4 * Cc);
  cvt_f32_to_f16<<<(Cc * W4 + 255) / 256, 256, 0, stream>>>(w2, w2_h, Cc * W4);
  transpose_x_f16<<<(MT * Cc) / 256, 256, 0, stream>>>(x, xT_h);

  // 2. QKV projection: [8192,256] x [768,256]^T -> [8192,768] f16 (+bias)
  gemm_wmma<<<dim3(C3 / 64, MT / 128), 128, 0, stream>>>(
      xT_h, wqkv_h, b_qkv, qkv_h, nullptr, MT, C3, Cc, MODE_F16_BIAS);

  // 3. V transpose for the PV matmul B-fragments
  transpose_v_f16<<<(Bn * Hh * DHd * Ll) / 256, 256, 0, stream>>>(qkv_h, vT_h);

  // 4. flash attention (softmax fused; scores never hit HBM)
  attn_flash<<<dim3(Ll / 16, Hh, Bn), 32, 0, stream>>>(qkv_h, vT_h, attn_h);

  // 5. out projection -> f32 [B,L,C]
  gemm_wmma<<<dim3(Cc / 64, MT / 128), 128, 0, stream>>>(
      attn_h, wo_h, b_o, hmha, nullptr, MT, Cc, Cc, MODE_F32_BIAS);

  // 6. GroupNorm(num_groups=1) over (C,L) per batch
  gn_zero<<<1, 32, 0, stream>>>(stats);
  gn_reduce<<<dim3(64, Bn), 256, 0, stream>>>(hmha, stats);
  gn_apply<<<(MT * Cc) / 256, 256, 0, stream>>>(hmha, stats, gn_g, gn_b, gn_h);

  // 7. MLP up-proj + exact GELU -> f16 [B,L,1024]
  gemm_wmma<<<dim3(W4 / 64, MT / 128), 128, 0, stream>>>(
      gn_h, w1_h, nullptr, h1g, nullptr, MT, W4, Cc, MODE_GELU_F16);

  // 8. MLP down-proj + residual add, store transposed to [B,C,L] f32
  gemm_wmma<<<dim3(Cc / 64, MT / 128), 128, 0, stream>>>(
      h1g, w2_h, nullptr, out, x, MT, Cc, W4, MODE_RESID_TR);

  (void)in_sizes; (void)n_in; (void)out_size; (void)ws_size;
}
